// LlamaMultiHeadAttention_5669356830738
// MI455X (gfx1250) — matrix operand order repaired and checked by reading, not compiled, not run on hardware
//
#include <hip/hip_runtime.h>
#include <hip/hip_bf16.h>

// Problem constants (from reference)
#define B_  2
#define S_  2048
#define D_  2048
#define H_  16
#define DH_ 128

typedef __attribute__((ext_vector_type(16))) __bf16 v16bf;
typedef __attribute__((ext_vector_type(8)))  __bf16 v8bf;
typedef __attribute__((ext_vector_type(4)))  __bf16 v4bf;
typedef __attribute__((ext_vector_type(8)))  float  v8f;
typedef __attribute__((ext_vector_type(4)))  float  v4f;
typedef __attribute__((ext_vector_type(4)))  unsigned int v4u;
typedef __attribute__((ext_vector_type(8)))  int    v8i;
typedef __attribute__((ext_vector_type(4)))  int    v4i;

static __device__ __forceinline__ v16bf pack16(v8bf lo, v8bf hi) {
  v16bf r;
#pragma unroll
  for (int i = 0; i < 8; ++i) { r[i] = lo[i]; r[i + 8] = hi[i]; }
  return r;
}

static __device__ __forceinline__ v8f zero8() {
  v8f z;
#pragma unroll
  for (int i = 0; i < 8; ++i) z[i] = 0.0f;
  return z;
}

static __device__ __forceinline__ v4bf cvt4(v4f f) {
  v4bf r;
#pragma unroll
  for (int i = 0; i < 4; ++i) r[i] = (__bf16)f[i];
  return r;
}

static __device__ __forceinline__ v8f wmma_bf16(v16bf a, v16bf b, v8f c) {
  // D = A(16x32 bf16) * B(32x16 bf16) + C(16x16 f32)
  return __builtin_amdgcn_wmma_f32_16x16x32_bf16(false, a, false, b, (short)0, c,
                                                 false, false);
}

// ---------------------------------------------------------------------------
// TDM: async 2D-tile DMA global(bf16) -> LDS via the Tensor Data Mover.
// Tile: 128 rows x 32 bf16 elements, row stride D_ elements; packed row-major
// into LDS (matches the As[m][32] layout). Descriptor per CDNA5 ISA sec 8.
// ---------------------------------------------------------------------------
static __device__ __forceinline__ void tdm_load_tile128x32_bf16(
    const __bf16* gbase, unsigned int lds_byte_off) {
  unsigned long long ga = (unsigned long long)(uintptr_t)gbase;
  v4u g0;
  g0[0] = 1u;                                   // count=1 (valid user D#)
  g0[1] = lds_byte_off;                         // lds_addr (bytes)
  g0[2] = (unsigned int)(ga & 0xFFFFFFFFu);     // global_addr[31:0]
  g0[3] = (unsigned int)((ga >> 32) & 0x01FFFFFFu) | 0x80000000u; // [56:32]|type=2
  v8i g1;
  g1[0] = (int)(1u << 16);                      // data_size=1 (2B), wg_mask=0
  g1[1] = (int)(((unsigned)D_ & 0xFFFFu) << 16);   // tensor_dim0 = 2048 (lo16)
  g1[2] = (int)((4096u & 0xFFFFu) << 16);          // dim0 hi16=0 | tensor_dim1 lo16
  g1[3] = (int)(32u << 16);                        // dim1 hi16=0 | tile_dim0 = 32
  g1[4] = (int)128u;                               // tile_dim1 = 128, tile_dim2=0
  g1[5] = (int)(unsigned)D_;                       // tensor_dim0_stride = 2048
  g1[6] = 0;                                       // stride hi / dim1_stride lo
  g1[7] = 0;                                       // dim1_stride hi
  v4i gz;
  gz[0] = 0; gz[1] = 0; gz[2] = 0; gz[3] = 0;      // groups 2/3 unused (2D)
#if defined(__clang_major__) && (__clang_major__ >= 23)
  v8i gz8;
#pragma unroll
  for (int i = 0; i < 8; ++i) gz8[i] = 0;
  __builtin_amdgcn_tensor_load_to_lds(g0, g1, gz, gz, gz8, 0);
#else
  __builtin_amdgcn_tensor_load_to_lds(g0, g1, gz, gz, 0);
#endif
}

// ---------------------------------------------------------------------------
// Kernel 0: RoPE cos/sin tables [S][DH/2] (1 MB total, stays in L2)
// ---------------------------------------------------------------------------
__global__ __launch_bounds__(256)
void rope_tables_kernel(float* __restrict__ cosT, float* __restrict__ sinT) {
  int idx = blockIdx.x * 256 + threadIdx.x;  // S_*64 entries
  int s = idx >> 6, k = idx & 63;
  // freq = 10000^{-2k/128} = exp(-k/64 * ln(10000))
  float freq = __expf(-(float)k * (9.210340371976184f / 64.0f));
  float sn, cs;
  __sincosf((float)s * freq, &sn, &cs);
  cosT[idx] = cs;
  sinT[idx] = sn;
}

// ---------------------------------------------------------------------------
// Kernel 1: fused Q/K/V projection GEMM (x[4096,2048] @ W[2048,2048]) with
// RoPE fused into the epilogue for Q and K. blockIdx.z selects the matrix.
//   Q -> Qb [bh][s][dh]  (bf16, RoPE applied)
//   K -> Kb [bh][s][dh]  (bf16, RoPE applied)
//   V -> Vt [bh][dh][s]  (bf16, transposed for the PV WMMA B-fragment)
// Double-buffered LDS with register prefetch; vectorized staging.
// ---------------------------------------------------------------------------
__global__ __launch_bounds__(256)
void qkv_rope_kernel(const float* __restrict__ x,
                     const float* __restrict__ wq,
                     const float* __restrict__ wk,
                     const float* __restrict__ wv,
                     const float* __restrict__ cosT,
                     const float* __restrict__ sinT,
                     __bf16* __restrict__ Qb,
                     __bf16* __restrict__ Kb,
                     __bf16* __restrict__ Vt) {
  __shared__ __align__(16) __bf16 As[2][128 * 32];  // [m][k] row-major
  __shared__ __align__(16) __bf16 Bs[2][128 * 32];  // [n][k] (B^T)

  const int tid  = threadIdx.x;
  const int lane = tid & 31, wid = tid >> 5;
  const int half = lane >> 4, l16 = lane & 15;
  const int waveM = wid >> 2, waveN = wid & 3;  // 2x4 wave grid
  const int mat = blockIdx.z;
  const float* W = (mat == 0) ? wq : (mat == 1) ? wk : wv;
  const int rowBase = blockIdx.x * 128;
  const int colBase = blockIdx.y * 128;

  v8f acc[4][2];
#pragma unroll
  for (int mt = 0; mt < 4; ++mt)
#pragma unroll
    for (int nt = 0; nt < 2; ++nt) acc[mt][nt] = zero8();

  // ---- prologue: stage k-block 0 into buffer 0 ----
#pragma unroll
  for (int j = 0; j < 4; ++j) {
    int i = tid + j * 256;
    int rA = i >> 3, cA = (i & 7) * 4;
    v4f a = *(const v4f*)&x[(size_t)(rowBase + rA) * D_ + cA];
    *(v4bf*)&As[0][rA * 32 + cA] = cvt4(a);
    int n = i & 127, k4 = (i >> 7) * 4;
    v4f b;
#pragma unroll
    for (int jj = 0; jj < 4; ++jj)
      b[jj] = W[(size_t)(k4 + jj) * D_ + colBase + n];
    *(v4bf*)&Bs[0][n * 32 + k4] = cvt4(b);
  }
  __syncthreads();

  for (int kk = 0; kk < 64; ++kk) {
    const int cur = kk & 1;
    const bool more = (kk + 1) < 64;

    // ---- issue global loads for next tile into registers ----
    v4f pa[4], pb[4];
    if (more) {
      const int k0 = (kk + 1) * 32;
#pragma unroll
      for (int j = 0; j < 4; ++j) {
        int i = tid + j * 256;
        int rA = i >> 3, cA = (i & 7) * 4;
        pa[j] = *(const v4f*)&x[(size_t)(rowBase + rA) * D_ + k0 + cA];
        int n = i & 127, k4 = (i >> 7) * 4;
#pragma unroll
        for (int jj = 0; jj < 4; ++jj)
          pb[j][jj] = W[(size_t)(k0 + k4 + jj) * D_ + colBase + n];
      }
      if (kk + 2 < 64) {  // L2 prefetch two tiles ahead
        const int kp = (kk + 2) * 32;
        __builtin_prefetch(&x[(size_t)(rowBase + (tid >> 3)) * D_ + kp], 0, 1);
        __builtin_prefetch(&W[(size_t)(kp + ((tid >> 7) * 4)) * D_ + colBase + (tid & 127)], 0, 1);
      }
    }

    // ---- WMMA on current buffer ----
    v16bf afr[4];
#pragma unroll
    for (int mt = 0; mt < 4; ++mt) {
      const __bf16* ap = &As[cur][(waveM * 64 + mt * 16 + l16) * 32];
      afr[mt] = pack16(*(const v8bf*)(ap + half * 16),
                       *(const v8bf*)(ap + half * 16 + 8));
    }
#pragma unroll
    for (int nt = 0; nt < 2; ++nt) {
      const __bf16* bp = &Bs[cur][(waveN * 32 + nt * 16 + l16) * 32];
      v16bf bfr = pack16(*(const v8bf*)(bp + half * 16),
                         *(const v8bf*)(bp + half * 16 + 8));
#pragma unroll
      for (int mt = 0; mt < 4; ++mt)
        acc[mt][nt] = wmma_bf16(afr[mt], bfr, acc[mt][nt]);
    }

    // ---- convert + store prefetched tile, one barrier/iter ----
    if (more) {
      const int nxt = cur ^ 1;
#pragma unroll
      for (int j = 0; j < 4; ++j) {
        int i = tid + j * 256;
        int rA = i >> 3, cA = (i & 7) * 4;
        *(v4bf*)&As[nxt][rA * 32 + cA] = cvt4(pa[j]);
        int n = i & 127, k4 = (i >> 7) * 4;
        *(v4bf*)&Bs[nxt][n * 32 + k4] = cvt4(pb[j]);
      }
      __syncthreads();
    }
  }

  // ---- epilogue: C layout M = v + 8*half, N = l16. RoPE pair partner
  //      (col^1) lives in lane^1 -> one shfl_xor; angles from tables. ----
#pragma unroll
  for (int mt = 0; mt < 4; ++mt) {
#pragma unroll
    for (int nt = 0; nt < 2; ++nt) {
#pragma unroll
      for (int v = 0; v < 8; ++v) {
        int r = rowBase + waveM * 64 + mt * 16 + v + half * 8;  // b*S + s
        int c = colBase + waveN * 32 + nt * 16 + l16;           // h*DH + dh
        int bb = r >> 11, s = r & (S_ - 1);
        int h = c >> 7, dh = c & (DH_ - 1);
        int bh = bb * H_ + h;
        float val = acc[mt][nt][v];
        if (mat < 2) {
          float partner = __shfl_xor(val, 1, 32);
          float cs = cosT[s * 64 + (dh >> 1)];
          float sn = sinT[s * 64 + (dh >> 1)];
          float out = (dh & 1) ? (val * cs + partner * sn)   // odd: xe*sin+xo*cos
                               : (val * cs - partner * sn);  // even: xe*cos-xo*sin
          __bf16* dst = (mat == 0) ? Qb : Kb;
          dst[((size_t)bh * S_ + s) * DH_ + dh] = (__bf16)out;
        } else {
          Vt[((size_t)bh * DH_ + dh) * S_ + s] = (__bf16)val;
        }
      }
    }
  }
}

// ---------------------------------------------------------------------------
// Kernel 2: causal flash attention, one wave per 16 query rows.
// Scores via 4x wmma (dh=128), online softmax with 16-lane shfl reductions,
// P transposed C->A layout through wave-private LDS, PV via 8x wmma vs Vt.
// ---------------------------------------------------------------------------
__global__ __launch_bounds__(256)
void flash_attn_kernel(const __bf16* __restrict__ Qb,
                       const __bf16* __restrict__ Kb,
                       const __bf16* __restrict__ Vt,
                       __bf16* __restrict__ Ob) {  // [b][s][h][dh]
  __shared__ __align__(16) __bf16 Pl[8][16][32];   // per-wave P tile

  const int tid  = threadIdx.x;
  const int lane = tid & 31, wid = tid >> 5;
  const int half = lane >> 4, l16 = lane & 15;
  const int bh = blockIdx.y;
  const int bb = bh / H_, h = bh % H_;
  const int q0 = blockIdx.x * 128 + wid * 16;
  const float scale = 0.08838834764831845f;  // 1/sqrt(128)

  // Preload Q A-fragments (4 dh-chunks of 32)
  v16bf aQ[4];
  {
    const __bf16* qp = Qb + ((size_t)bh * S_ + (q0 + l16)) * DH_;
#pragma unroll
    for (int dc = 0; dc < 4; ++dc) {
      const __bf16* p = qp + dc * 32;
      aQ[dc] = pack16(*(const v8bf*)(p + half * 16),
                      *(const v8bf*)(p + half * 16 + 8));
    }
  }

  v8f Oacc[8];
#pragma unroll
  for (int t = 0; t < 8; ++t) Oacc[t] = zero8();
  float mrow[8], lrow[8];
#pragma unroll
  for (int v = 0; v < 8; ++v) { mrow[v] = -3.0e38f; lrow[v] = 0.0f; }

  const int kmax = q0 + 16;            // exclusive causal key bound
  const int nkb = (kmax + 31) >> 5;    // 32-key blocks

  for (int kb = 0; kb < nkb; ++kb) {
    // ---- S = Q K^T for a 16x32 key slab (two 16x16 N-tiles) ----
    v8f s0 = zero8(), s1 = zero8();
#pragma unroll
    for (int dc = 0; dc < 4; ++dc) {
      const __bf16* kp0 =
          Kb + ((size_t)bh * S_ + kb * 32 + l16) * DH_ + dc * 32 + half * 16;
      const __bf16* kp1 =
          Kb + ((size_t)bh * S_ + kb * 32 + 16 + l16) * DH_ + dc * 32 + half * 16;
      v16bf b0 = pack16(*(const v8bf*)kp0, *(const v8bf*)(kp0 + 8));
      v16bf b1 = pack16(*(const v8bf*)kp1, *(const v8bf*)(kp1 + 8));
      s0 = wmma_bf16(aQ[dc], b0, s0);
      s1 = wmma_bf16(aQ[dc], b1, s1);
    }

    // ---- online softmax (per C row: M = v + 8*half) ----
#pragma unroll
    for (int v = 0; v < 8; ++v) {
      int qrow = q0 + v + half * 8;
      int key0 = kb * 32 + l16;
      int key1 = key0 + 16;
      float x0 = (key0 <= qrow) ? s0[v] * scale : -3.0e38f;
      float x1 = (key1 <= qrow) ? s1[v] * scale : -3.0e38f;
      float mx = fmaxf(x0, x1);
#pragma unroll
      for (int mm = 1; mm < 16; mm <<= 1) mx = fmaxf(mx, __shfl_xor(mx, mm, 32));
      float mnew = fmaxf(mrow[v], mx);
      float corr = __expf(mrow[v] - mnew);
      float p0 = __expf(x0 - mnew);
      float p1 = __expf(x1 - mnew);
      float psum = p0 + p1;
#pragma unroll
      for (int mm = 1; mm < 16; mm <<= 1) psum += __shfl_xor(psum, mm, 32);
      lrow[v] = lrow[v] * corr + psum;
      mrow[v] = mnew;
#pragma unroll
      for (int t = 0; t < 8; ++t) Oacc[t][v] *= corr;
      // C-layout -> LDS (row-major 16x32) for A-fragment reload
      Pl[wid][v + half * 8][l16]      = (__bf16)p0;
      Pl[wid][v + half * 8][16 + l16] = (__bf16)p1;
    }

    // reload P as a 16x32 A fragment (wave-private; DS ops in-order per wave)
    const __bf16* pp = &Pl[wid][l16][0];
    v16bf aP = pack16(*(const v8bf*)(pp + half * 16),
                      *(const v8bf*)(pp + half * 16 + 8));

    // ---- O += P V : 8 dh-tiles of 16, Vt row = contiguous keys ----
#pragma unroll
    for (int t = 0; t < 8; ++t) {
      const __bf16* vp =
          Vt + ((size_t)bh * DH_ + t * 16 + l16) * S_ + kb * 32 + half * 16;
      v16bf bV = pack16(*(const v8bf*)vp, *(const v8bf*)(vp + 8));
      Oacc[t] = wmma_bf16(aP, bV, Oacc[t]);
    }
  }

  // normalize + store bf16 [b][s][h][dh]
#pragma unroll
  for (int t = 0; t < 8; ++t) {
#pragma unroll
    for (int v = 0; v < 8; ++v) {
      int qrow = q0 + v + half * 8;
      int dh = t * 16 + l16;
      float o = Oacc[t][v] / lrow[v];
      Ob[(((size_t)bb * S_ + qrow) * H_ + h) * DH_ + dh] = (__bf16)o;
    }
  }
}

// ---------------------------------------------------------------------------
// Kernel 3: output projection, attn[4096,2048](bf16) @ Wo[2048,2048](f32->bf16)
// -> d_out fp32. A tile staged via TDM async tensor DMA (TENSORcnt-fenced);
// Wo staged via register path (needs f32->bf16 convert + transpose).
// ---------------------------------------------------------------------------
__global__ __launch_bounds__(256)
void out_proj_kernel(const __bf16* __restrict__ A,
                     const float* __restrict__ Wo,
                     float* __restrict__ out) {
  __shared__ __align__(16) __bf16 As[2][128 * 32];
  __shared__ __align__(16) __bf16 Bs[2][128 * 32];

  const int tid  = threadIdx.x;
  const int lane = tid & 31, wid = tid >> 5;
  const int half = lane >> 4, l16 = lane & 15;
  const int waveM = wid >> 2, waveN = wid & 3;
  const int rowBase = blockIdx.x * 128;
  const int colBase = blockIdx.y * 128;

  v8f acc[4][2];
#pragma unroll
  for (int mt = 0; mt < 4; ++mt)
#pragma unroll
    for (int nt = 0; nt < 2; ++nt) acc[mt][nt] = zero8();

  // ---- prologue stage (buffer 0): A via TDM, B via registers ----
  if (wid == 0) {
    tdm_load_tile128x32_bf16(&A[(size_t)rowBase * D_],
                             (unsigned int)(uintptr_t)&As[0][0]);
  }
#pragma unroll
  for (int j = 0; j < 4; ++j) {
    int i = tid + j * 256;
    int n = i & 127, k4 = (i >> 7) * 4;
    v4f b;
#pragma unroll
    for (int jj = 0; jj < 4; ++jj)
      b[jj] = Wo[(size_t)(k4 + jj) * D_ + colBase + n];
    *(v4bf*)&Bs[0][n * 32 + k4] = cvt4(b);
  }
  if (wid == 0) __builtin_amdgcn_s_wait_tensorcnt(0);
  __syncthreads();

  for (int kk = 0; kk < 64; ++kk) {
    const int cur = kk & 1;
    const int nxt = cur ^ 1;
    const bool more = (kk + 1) < 64;

    // ---- kick off next A tile DMA + issue next B loads ----
    v4f pb[4];
    if (more) {
      const int k0 = (kk + 1) * 32;
      if (wid == 0) {
        tdm_load_tile128x32_bf16(&A[(size_t)rowBase * D_ + k0],
                                 (unsigned int)(uintptr_t)&As[nxt][0]);
      }
#pragma unroll
      for (int j = 0; j < 4; ++j) {
        int i = tid + j * 256;
        int n = i & 127, k4 = (i >> 7) * 4;
#pragma unroll
        for (int jj = 0; jj < 4; ++jj)
          pb[j][jj] = Wo[(size_t)(k0 + k4 + jj) * D_ + colBase + n];
      }
      if (kk + 2 < 64) {
        const int kp = (kk + 2) * 32;
        __builtin_prefetch(&Wo[(size_t)(kp + ((tid >> 7) * 4)) * D_ + colBase + (tid & 127)], 0, 1);
      }
    }

    // ---- WMMA on current buffer ----
    v16bf afr[4];
#pragma unroll
    for (int mt = 0; mt < 4; ++mt) {
      const __bf16* ap = &As[cur][(waveM * 64 + mt * 16 + l16) * 32];
      afr[mt] = pack16(*(const v8bf*)(ap + half * 16),
                       *(const v8bf*)(ap + half * 16 + 8));
    }
#pragma unroll
    for (int nt = 0; nt < 2; ++nt) {
      const __bf16* bp = &Bs[cur][(waveN * 32 + nt * 16 + l16) * 32];
      v16bf bfr = pack16(*(const v8bf*)(bp + half * 16),
                         *(const v8bf*)(bp + half * 16 + 8));
#pragma unroll
      for (int mt = 0; mt < 4; ++mt)
        acc[mt][nt] = wmma_bf16(afr[mt], bfr, acc[mt][nt]);
    }

    // ---- store B, fence the TDM copy, one barrier/iter ----
    if (more) {
#pragma unroll
      for (int j = 0; j < 4; ++j) {
        int i = tid + j * 256;
        int n = i & 127, k4 = (i >> 7) * 4;
        *(v4bf*)&Bs[nxt][n * 32 + k4] = cvt4(pb[j]);
      }
      if (wid == 0) __builtin_amdgcn_s_wait_tensorcnt(0);
      __syncthreads();
    }
  }

#pragma unroll
  for (int mt = 0; mt < 4; ++mt) {
#pragma unroll
    for (int nt = 0; nt < 2; ++nt) {
#pragma unroll
      for (int v = 0; v < 8; ++v) {
        int r = rowBase + waveM * 64 + mt * 16 + v + half * 8;
        int c = colBase + waveN * 32 + nt * 16 + l16;
        out[(size_t)r * D_ + c] = acc[mt][nt][v];
      }
    }
  }
}

// ---------------------------------------------------------------------------
extern "C" void kernel_launch(void* const* d_in, const int* in_sizes, int n_in,
                              void* d_out, int out_size, void* d_ws,
                              size_t ws_size, hipStream_t stream) {
  const float* x  = (const float*)d_in[0];
  const float* wq = (const float*)d_in[1];
  const float* wk = (const float*)d_in[2];
  const float* wv = (const float*)d_in[3];
  const float* wo = (const float*)d_in[4];
  float* out = (float*)d_out;

  const size_t elems = (size_t)B_ * H_ * S_ * DH_;  // 8M bf16 elems per buffer
  __bf16* Qb = (__bf16*)d_ws;
  __bf16* Kb = Qb + elems;
  __bf16* Vt = Kb + elems;
  __bf16* Ob = Vt + elems;                          // 64 MB of bf16 buffers
  float* cosT = (float*)(Ob + elems);               // [S][64]
  float* sinT = cosT + (size_t)S_ * 64;             // [S][64]

  dim3 blk(256);
  // 0) RoPE tables (1 MB, L2-resident)
  rope_tables_kernel<<<dim3((S_ * 64) / 256), blk, 0, stream>>>(cosT, sinT);
  // 1) QKV projections + fused RoPE (+ V transpose)
  qkv_rope_kernel<<<dim3(32, 16, 3), blk, 0, stream>>>(x, wq, wk, wv, cosT, sinT,
                                                       Qb, Kb, Vt);
  // 2) causal flash attention per (b,h)
  flash_attn_kernel<<<dim3(16, 32), blk, 0, stream>>>(Qb, Kb, Vt, Ob);
  // 3) output projection -> fp32
  out_proj_kernel<<<dim3(32, 16), blk, 0, stream>>>(Ob, wo, out);
}
